// gMLPSAGPoolGraphClassification2_38809324487501
// MI455X (gfx1250) — matrix operand order repaired and checked by reading, not compiled, not run on hardware
//
#include <hip/hip_runtime.h>

typedef __attribute__((ext_vector_type(16))) _Float16 v16h;
typedef __attribute__((ext_vector_type(8)))  _Float16 v8h;
typedef __attribute__((ext_vector_type(8)))  float    v8f;
typedef __attribute__((ext_vector_type(4)))  int      v4i;

#if defined(__gfx1250__) && __has_builtin(__builtin_amdgcn_global_load_async_to_lds_b128)
#define HAVE_ASYNC_LDS 1
typedef __attribute__((address_space(1))) v4i as1_v4i;   // global memory v4i
typedef __attribute__((address_space(3))) v4i as3_v4i;   // LDS v4i
#else
#define HAVE_ASYNC_LDS 0
#endif

namespace {

constexpr int kB    = 32;
constexpr int kN0   = 512;
constexpr int kE    = 262144;
constexpr int kNF   = 128;
constexpr int kHID  = 256;
constexpr int kFFN  = 512;
constexpr int kNCLS = 10;
constexpr int kNL   = 3;
constexpr int kNH   = 3;
constexpr int kJK   = 2 * kNL * kHID;   // 1536

enum { EPI_BIAS = 0, EPI_GELU = 1, EPI_RES = 2, EPI_RELU = 3 };

__device__ __forceinline__ float gelu_f(float x) {
  float x3 = x * x * x;
  return 0.5f * x * (1.0f + tanhf(0.7978845608028654f * (x + 0.044715f * x3)));
}

// ---------------------------------------------------------------------------
// WMMA GEMM: C[M,N] = A[M,K] * B[K,N] (+bias, epilogue). B passed transposed
// as Bt[N,K] f16. Block = 4 waves; block tile = 64(M) x 64(N); each wave owns
// a 16x64 strip (4 accumulators, A fragment reused 4x). The 64x32 B tile is
// staged per K-step into double-buffered LDS via GLOBAL_LOAD_ASYNC_TO_LDS_B128
// (ASYNCcnt) and consumed with ds_load_b128.
// ---------------------------------------------------------------------------
template <int EPI>
__global__ __launch_bounds__(128)
void wmma_gemm_f16(const _Float16* __restrict__ A,   // [M,K] row-major f16
                   const _Float16* __restrict__ Bt,  // [N,K] row-major (B^T)
                   const float* __restrict__ bias,   // [N] or null
                   const float* resid,               // [M,N] or null (may alias C)
                   float* C,                         // [M,N] fp32
                   _Float16* C16,                    // optional f16 mirror
                   int M, int N, int K)
{
  __shared__ _Float16 ldsB[2][64 * 32];   // [buf][n_in_tile*32 + k_in_tile], 2x4KB

  const int tid  = threadIdx.x;
  const int lane = tid & 31;
  const int wave = tid >> 5;
  const int nblk = blockIdx.x * 64;
  const int m0   = blockIdx.y * 64 + wave * 16;
  const int r    = lane & 15;   // row (A) / col (B) within a 16x16 tile
  const int hh   = lane >> 4;   // K-half selector per 16-bit VGPR layouts

  const int arow = min(m0 + r, M - 1);    // clamp for ragged M (classifier GEMMs)
  const _Float16* Arow = A + (size_t)arow * K;

  // Stage one 64(N) x 32(K) B tile (4KB) into LDS: 128 threads x 2 chunks x 16B.
  auto stage = [&](int buf, int k0) {
#pragma unroll
    for (int pass = 0; pass < 2; ++pass) {
      const int c   = pass * 128 + tid;
      const int row = c >> 2;             // 0..63 (n within tile)
      const int seg = c & 3;              // 16B segment within 64B row
      const _Float16* g = Bt + (size_t)(nblk + row) * K + k0 + seg * 8;
      _Float16*       l = &ldsB[buf][row * 32 + seg * 8];
#if HAVE_ASYNC_LDS
      as1_v4i* gp = (as1_v4i*)(unsigned long long)g;
      as3_v4i* lp = (as3_v4i*)(unsigned)(unsigned long long)l;   // low 32b = LDS offset
      __builtin_amdgcn_global_load_async_to_lds_b128(gp, lp, 0, 0);
#else
      *(float4*)l = *(const float4*)g;
#endif
    }
  };

  auto stage_fence = [&]() {
#if HAVE_ASYNC_LDS
#if __has_builtin(__builtin_amdgcn_s_wait_asynccnt)
    __builtin_amdgcn_s_wait_asynccnt(0);
#else
    asm volatile("s_wait_asynccnt 0x0" ::: "memory");
#endif
#endif
    __syncthreads();
  };

  v8f acc[4] = {};

  stage(0, 0);
  for (int k0 = 0; k0 < K; k0 += 32) {
    const int buf = (k0 >> 5) & 1;
    stage_fence();                        // drain ASYNCcnt + make tile visible
    if (k0 + 32 < K) {
      stage(buf ^ 1, k0 + 32);            // overlap next tile with this K-step
      __builtin_prefetch(Arow + k0 + 64, 0, 0);   // global_prefetch_b8
    }

    // A 16x32 fragment: lanes 0-15 hold K 0..7/16..23, lanes 16-31 K 8..15/24..31
    v8h a_lo = *(const v8h*)(Arow + k0 + hh * 8);
    v8h a_hi = *(const v8h*)(Arow + k0 + 16 + hh * 8);
    v16h av;
#pragma unroll
    for (int i = 0; i < 8; ++i) { av[i] = a_lo[i]; av[8 + i] = a_hi[i]; }

#pragma unroll
    for (int t = 0; t < 4; ++t) {
      // B 32x16 fragment from LDS: lanes 0-15 K 0..7/16..23, lanes 16-31 K 8..15/24..31
      const _Float16* lb = &ldsB[buf][(t * 16 + r) * 32 + hh * 8];
      v8h b_lo = *(const v8h*)(lb);
      v8h b_hi = *(const v8h*)(lb + 16);
      v16h bv;
#pragma unroll
      for (int i = 0; i < 8; ++i) { bv[i] = b_lo[i]; bv[8 + i] = b_hi[i]; }
      acc[t] = __builtin_amdgcn_wmma_f32_16x16x32_f16(false, av, false, bv,
                                                      (short)0, acc[t], false, false);
    }
  }

  // C layout: VGPR j -> row m0 + hh*8 + j, col ntile + (lane&15)
#pragma unroll
  for (int t = 0; t < 4; ++t) {
    const int col = nblk + t * 16 + r;
    const float bb = bias ? bias[col] : 0.0f;
#pragma unroll
    for (int j = 0; j < 8; ++j) {
      const int row = m0 + hh * 8 + j;
      if (row < M) {
        float v = acc[t][j] + bb;
        if (EPI == EPI_GELU)      v = gelu_f(v);
        else if (EPI == EPI_RES)  v += resid[(size_t)row * N + col];
        else if (EPI == EPI_RELU) v = fmaxf(v, 0.0f);
        C[(size_t)row * N + col] = v;
        if (C16) C16[(size_t)row * N + col] = (_Float16)v;
      }
    }
  }
}

// ---------------------------------------------------------------------------
// LayerNorm over 256 features; one wave32 per row; writes f16 or f32.
// ---------------------------------------------------------------------------
__global__ void layernorm_256(const float* __restrict__ X, int xstride,
                              const float* __restrict__ gam,
                              const float* __restrict__ bet,
                              _Float16* __restrict__ Y16, float* __restrict__ Y32,
                              int nrows)
{
  const int lane = threadIdx.x & 31;
  const int wave = threadIdx.x >> 5;
  const int row  = blockIdx.x * 8 + wave;
  if (row >= nrows) return;
  const float* xr = X + (size_t)row * xstride + lane * 8;
  float v[8];
  *(float4*)(v + 0) = *(const float4*)(xr + 0);
  *(float4*)(v + 4) = *(const float4*)(xr + 4);
  float s = 0.f, ss = 0.f;
#pragma unroll
  for (int i = 0; i < 8; ++i) { s += v[i]; ss += v[i] * v[i]; }
#pragma unroll
  for (int m = 16; m >= 1; m >>= 1) { s += __shfl_xor(s, m, 32); ss += __shfl_xor(ss, m, 32); }
  const float mean = s * (1.0f / 256.0f);
  const float var  = ss * (1.0f / 256.0f) - mean * mean;
  const float rstd = rsqrtf(var + 1e-5f);
  const int fb = lane * 8;
#pragma unroll
  for (int i = 0; i < 8; ++i) {
    const float y = (v[i] - mean) * rstd * gam[fb + i] + bet[fb + i];
    if (Y16) Y16[(size_t)row * 256 + fb + i] = (_Float16)y;
    else     Y32[(size_t)row * 256 + fb + i] = y;
  }
}

// Masked edge segment-sum: out[dst] += V[src] * emask. 64 threads/edge, float4.
__global__ void edge_agg_256(const int* __restrict__ src, const int* __restrict__ dst,
                             const float* __restrict__ emask,
                             const float* __restrict__ V, float* __restrict__ out)
{
  const int t = blockIdx.x * blockDim.x + threadIdx.x;
  const int e = t >> 6;
  if (e >= kE) return;
  const float m = emask[e];
  if (m == 0.0f) return;
  const int f = (t & 63) * 4;
  const float4 v = *(const float4*)(V + (size_t)src[e] * 256 + f);
  float* o = out + (size_t)dst[e] * 256 + f;
  atomicAdd(o + 0, v.x * m);
  atomicAdd(o + 1, v.y * m);
  atomicAdd(o + 2, v.z * m);
  atomicAdd(o + 3, v.w * m);
}

// g = u * (vagg * w_s + b_s), u = first half of T[n,512]; output f16 GEMM operand.
__global__ void gate_256(const float* __restrict__ T, const float* __restrict__ vagg,
                         const float* __restrict__ ws, const float* __restrict__ bs,
                         _Float16* __restrict__ G, int n)
{
  const int idx = blockIdx.x * blockDim.x + threadIdx.x;
  if (idx >= n * 256) return;
  const int i = idx >> 8, f = idx & 255;
  const float u = T[(size_t)i * 512 + f];
  const float v = vagg[idx] * ws[f] + bs[f];
  G[idx] = (_Float16)(u * v);
}

// Per-graph mean+max readout (graph rows stay contiguous through SAGPool).
__global__ void readout_256(const float* __restrict__ Y, float* __restrict__ feats,
                            int nper, int li)
{
  const int b = blockIdx.x;
  const int f = threadIdx.x;
  const float* p = Y + (size_t)b * nper * 256 + f;
  float s = 0.f, mx = -3.4028235e38f;
  for (int i = 0; i < nper; ++i) {
    const float v = p[(size_t)i * 256];
    s += v; mx = fmaxf(mx, v);
  }
  feats[(size_t)b * kJK + li * 256 + f]         = s / (float)nper;
  feats[(size_t)b * kJK + (kNL + li) * 256 + f] = mx;
}

__global__ void gelu_accum(float* __restrict__ jk, const float* __restrict__ feats, int n) {
  const int i = blockIdx.x * blockDim.x + threadIdx.x;
  if (i < n) jk[i] += gelu_f(feats[i]);
}

// GraphConv score: dot(agg, rel_w) + rel_b + dot(x, root_w) + root_b.
__global__ void score_node(const float* __restrict__ X, const float* __restrict__ agg,
                           const float* __restrict__ rootw, const float* __restrict__ rootb,
                           const float* __restrict__ relw, const float* __restrict__ relb,
                           float* __restrict__ score, int n)
{
  const int lane = threadIdx.x & 31;
  const int wave = threadIdx.x >> 5;
  const int i = blockIdx.x * 8 + wave;
  if (i >= n) return;
  float s = 0.f;
  for (int f = lane; f < 256; f += 32)
    s += agg[(size_t)i * 256 + f] * relw[f] + X[(size_t)i * 256 + f] * rootw[f];
#pragma unroll
  for (int m = 16; m >= 1; m >>= 1) s += __shfl_xor(s, m, 32);
  if (lane == 0) score[i] = s + relb[0] + rootb[0];
}

// Per-graph top-k via in-LDS bitonic sort (nper <= 512).
__global__ __launch_bounds__(512)
void topk_graph(const float* __restrict__ score, int* __restrict__ perm, int nper, int k)
{
  __shared__ float sv[512];
  __shared__ int   si[512];
  const int b = blockIdx.x, t = threadIdx.x;
  sv[t] = (t < nper) ? score[(size_t)b * nper + t] : -3.4028235e38f;
  si[t] = t;
  __syncthreads();
  for (int size = 2; size <= 512; size <<= 1) {
    for (int stride = size >> 1; stride > 0; stride >>= 1) {
      const int j = t ^ stride;
      if (j > t) {
        const bool up = ((t & size) == 0);          // descending segments
        const float a = sv[t], c = sv[j];
        if (up ? (a < c) : (a > c)) {
          sv[t] = c; sv[j] = a;
          const int tmp = si[t]; si[t] = si[j]; si[j] = tmp;
        }
      }
      __syncthreads();
    }
  }
  if (t < k) perm[(size_t)b * k + t] = b * nper + si[t];
}

__global__ void pool_gather(const float* __restrict__ X, const float* __restrict__ score,
                            const int* __restrict__ perm, float* __restrict__ Xn,
                            int* __restrict__ newpos, int nk)
{
  const int idx = blockIdx.x * blockDim.x + threadIdx.x;
  if (idx >= nk * 256) return;
  const int i = idx >> 8, f = idx & 255;
  const int old = perm[i];
  Xn[idx] = X[(size_t)old * 256 + f] * tanhf(score[old]);
  if (f == 0) newpos[old] = i;
}

__global__ void edge_remap(const int* __restrict__ src, const int* __restrict__ dst,
                           const float* __restrict__ emask, const int* __restrict__ newpos,
                           int* __restrict__ src2, int* __restrict__ dst2,
                           float* __restrict__ emask2)
{
  const int e = blockIdx.x * blockDim.x + threadIdx.x;
  if (e >= kE) return;
  const int ns = newpos[src[e]];
  const int nd = newpos[dst[e]];
  const bool valid = (ns >= 0) && (nd >= 0) && (emask[e] > 0.0f);
  src2[e] = valid ? ns : 0;
  dst2[e] = valid ? nd : 0;
  emask2[e] = valid ? 1.0f : 0.0f;
}

__global__ void final_linear(const float* __restrict__ Z, const float* __restrict__ W3,
                             const float* __restrict__ b3, float* __restrict__ out)
{
  const int t = blockIdx.x * blockDim.x + threadIdx.x;
  if (t >= kB * kNCLS) return;
  const int b = t / kNCLS, c = t % kNCLS;
  float s = b3[c];
  for (int k = 0; k < 256; ++k) s += Z[(size_t)b * 256 + k] * W3[(size_t)k * kNCLS + c];
  out[t] = s;
}

__global__ void fill_f32(float* p, float v, int n) {
  const int i = blockIdx.x * blockDim.x + threadIdx.x;
  if (i < n) p[i] = v;
}
__global__ void fill_i32(int* p, int v, int n) {
  const int i = blockIdx.x * blockDim.x + threadIdx.x;
  if (i < n) p[i] = v;
}
__global__ void cvt_f32_f16(const float* __restrict__ s, _Float16* __restrict__ d, int n) {
  const int i = blockIdx.x * blockDim.x + threadIdx.x;
  if (i < n) d[i] = (_Float16)s[i];
}
// fp32 W[K,N] -> f16 Wt[N,K] so GEMM B fragments are contiguous along K.
__global__ void transpose_f16(const float* __restrict__ W, _Float16* __restrict__ Wt,
                              int K, int N) {
  const int i = blockIdx.x * blockDim.x + threadIdx.x;
  if (i >= K * N) return;
  const int k = i / N, n = i % N;
  Wt[(size_t)n * K + k] = (_Float16)W[i];
}

} // namespace

extern "C" void kernel_launch(void* const* d_in, const int* in_sizes, int n_in,
                              void* d_out, int out_size, void* d_ws, size_t ws_size,
                              hipStream_t stream)
{
  (void)in_sizes; (void)n_in; (void)out_size; (void)ws_size;

  const float* x           = (const float*)d_in[0];
  const int*   edge_index  = (const int*)d_in[1];
  const float* emb_w       = (const float*)d_in[3];
  const float* emb_b       = (const float*)d_in[4];
  const float* ln1_g       = (const float*)d_in[5];
  const float* ln1_b       = (const float*)d_in[6];
  const float* w_in        = (const float*)d_in[7];
  const float* b_in        = (const float*)d_in[8];
  const float* ln2_g       = (const float*)d_in[9];
  const float* ln2_b       = (const float*)d_in[10];
  const float* w_s         = (const float*)d_in[11];
  const float* b_s         = (const float*)d_in[12];
  const float* w_out       = (const float*)d_in[13];
  const float* b_out       = (const float*)d_in[14];
  const float* pool_root_w = (const float*)d_in[15];
  const float* pool_root_b = (const float*)d_in[16];
  const float* pool_rel_w  = (const float*)d_in[17];
  const float* pool_rel_b  = (const float*)d_in[18];
  const float* w1          = (const float*)d_in[19];
  const float* b1          = (const float*)d_in[20];
  const float* w2          = (const float*)d_in[21];
  const float* b2          = (const float*)d_in[22];
  const float* w3          = (const float*)d_in[23];
  const float* b3          = (const float*)d_in[24];
  float* out               = (float*)d_out;

  char* base = (char*)d_ws;
  size_t off = 0;
  auto alloc = [&](size_t bytes) -> void* {
    void* p = base + off;
    off = (off + bytes + 255) & ~(size_t)255;
    return p;
  };

  const int nmax = kB * kN0;   // 16384
  _Float16* xf16  = (_Float16*)alloc((size_t)nmax * kNF * 2);
  _Float16* wembT = (_Float16*)alloc((size_t)kHID * kNF * 2);
  _Float16* winT  = (_Float16*)alloc((size_t)kNH * kNL * kFFN * kHID * 2);
  _Float16* woutT = (_Float16*)alloc((size_t)kNH * kNL * kHID * kHID * 2);
  _Float16* w1T   = (_Float16*)alloc((size_t)kHID * kJK * 2);
  _Float16* w2T   = (_Float16*)alloc((size_t)kHID * kHID * 2);
  _Float16* a16   = (_Float16*)alloc((size_t)nmax * kHID * 2);
  _Float16* g16   = (_Float16*)alloc((size_t)nmax * kHID * 2);
  _Float16* jk16  = (_Float16*)alloc((size_t)kB * kJK * 2);
  _Float16* z16a  = (_Float16*)alloc((size_t)kB * kHID * 2);
  float* hA     = (float*)alloc((size_t)nmax * kHID * 4);
  float* hB     = (float*)alloc((size_t)(nmax / 2) * kHID * 4);
  float* T      = (float*)alloc((size_t)nmax * kFFN * 4);
  float* v2     = (float*)alloc((size_t)nmax * kHID * 4);
  float* vagg   = (float*)alloc((size_t)nmax * kHID * 4);
  float* score  = (float*)alloc((size_t)nmax * 4);
  float* feats  = (float*)alloc((size_t)kB * kJK * 4);
  float* jk     = (float*)alloc((size_t)kB * kJK * 4);
  float* z1     = (float*)alloc((size_t)kB * kHID * 4);
  float* z2     = (float*)alloc((size_t)kB * kHID * 4);
  float* emask0 = (float*)alloc((size_t)kE * 4);
  float* emaskA = (float*)alloc((size_t)kE * 4);
  float* emaskB = (float*)alloc((size_t)kE * 4);
  int* srcA   = (int*)alloc((size_t)kE * 4);
  int* dstA   = (int*)alloc((size_t)kE * 4);
  int* srcB   = (int*)alloc((size_t)kE * 4);
  int* dstB   = (int*)alloc((size_t)kE * 4);
  int* newpos = (int*)alloc((size_t)nmax * 4);
  int* perm   = (int*)alloc((size_t)nmax * 4);

  const int TPB = 256;
  auto nb  = [](long long n, int t) { return (int)((n + t - 1) / t); };
  auto gy  = [](int M) { return (unsigned)((M + 63) / 64); };

  // --- one-time prep: f16 operands, transposed weights, masks, jk = 0 ---
  cvt_f32_f16<<<nb((long long)nmax * kNF, TPB), TPB, 0, stream>>>(x, xf16, nmax * kNF);
  transpose_f16<<<nb(kNF * kHID, TPB), TPB, 0, stream>>>(emb_w, wembT, kNF, kHID);
  for (int s9 = 0; s9 < kNH * kNL; ++s9) {
    transpose_f16<<<nb(kHID * kFFN, TPB), TPB, 0, stream>>>(
        w_in + (size_t)s9 * kHID * kFFN, winT + (size_t)s9 * kFFN * kHID, kHID, kFFN);
    transpose_f16<<<nb(kHID * kHID, TPB), TPB, 0, stream>>>(
        w_out + (size_t)s9 * kHID * kHID, woutT + (size_t)s9 * kHID * kHID, kHID, kHID);
  }
  transpose_f16<<<nb(kJK * kHID, TPB), TPB, 0, stream>>>(w1, w1T, kJK, kHID);
  transpose_f16<<<nb(kHID * kHID, TPB), TPB, 0, stream>>>(w2, w2T, kHID, kHID);
  fill_f32<<<nb(kE, TPB), TPB, 0, stream>>>(emask0, 1.0f, kE);
  fill_f32<<<nb(kB * kJK, TPB), TPB, 0, stream>>>(jk, 0.0f, kB * kJK);

  // --- embedding: h = x @ emb_w + emb_b ---
  wmma_gemm_f16<EPI_BIAS><<<dim3(kHID / 64, gy(nmax)), 128, 0, stream>>>(
      xf16, wembT, emb_b, nullptr, hA, nullptr, nmax, kHID, kNF);

  float* hcur = hA;
  const int* cs = edge_index;        // src row of edge_index
  const int* cd = edge_index + kE;   // dst row
  const float* cm = emask0;
  int nper = kN0;

  for (int hi = 0; hi < kNH; ++hi) {
    const int n = kB * nper;
    for (int li = 0; li < kNL; ++li) {
      const int s9 = hi * kNL + li;
      layernorm_256<<<(n + 7) / 8, 256, 0, stream>>>(
          hcur, kHID, ln1_g + (size_t)s9 * kHID, ln1_b + (size_t)s9 * kHID, a16, nullptr, n);
      wmma_gemm_f16<EPI_GELU><<<dim3(kFFN / 64, gy(n)), 128, 0, stream>>>(
          a16, winT + (size_t)s9 * kFFN * kHID, b_in + (size_t)s9 * kFFN,
          nullptr, T, nullptr, n, kFFN, kHID);
      layernorm_256<<<(n + 7) / 8, 256, 0, stream>>>(
          T + kHID, kFFN, ln2_g + (size_t)s9 * kHID, ln2_b + (size_t)s9 * kHID, nullptr, v2, n);
      fill_f32<<<nb((long long)n * kHID, TPB), TPB, 0, stream>>>(vagg, 0.0f, n * kHID);
      edge_agg_256<<<nb((long long)kE * 64, TPB), TPB, 0, stream>>>(cs, cd, cm, v2, vagg);
      gate_256<<<nb((long long)n * kHID, TPB), TPB, 0, stream>>>(
          T, vagg, w_s + (size_t)s9 * kHID, b_s + (size_t)s9 * kHID, g16, n);
      wmma_gemm_f16<EPI_RES><<<dim3(kHID / 64, gy(n)), 128, 0, stream>>>(
          g16, woutT + (size_t)s9 * kHID * kHID, b_out + (size_t)s9 * kHID,
          hcur, hcur, nullptr, n, kHID, kHID);
      readout_256<<<kB, kHID, 0, stream>>>(hcur, feats, nper, li);
    }
    gelu_accum<<<nb(kB * kJK, TPB), TPB, 0, stream>>>(jk, feats, kB * kJK);

    if (hi < kNH - 1) {
      fill_f32<<<nb((long long)n * kHID, TPB), TPB, 0, stream>>>(vagg, 0.0f, n * kHID);
      edge_agg_256<<<nb((long long)kE * 64, TPB), TPB, 0, stream>>>(cs, cd, cm, hcur, vagg);
      score_node<<<(n + 7) / 8, 256, 0, stream>>>(
          hcur, vagg, pool_root_w + (size_t)hi * kHID, pool_root_b + hi,
          pool_rel_w + (size_t)hi * kHID, pool_rel_b + hi, score, n);
      const int k = nper / 2;
      topk_graph<<<kB, 512, 0, stream>>>(score, perm, nper, k);
      fill_i32<<<nb(n, TPB), TPB, 0, stream>>>(newpos, -1, n);
      float* hnext = (hi == 0) ? hB : hA;
      pool_gather<<<nb((long long)kB * k * kHID, TPB), TPB, 0, stream>>>(
          hcur, score, perm, hnext, newpos, kB * k);
      int*   ns_ = (hi == 0) ? srcA : srcB;
      int*   nd_ = (hi == 0) ? dstA : dstB;
      float* nm_ = (hi == 0) ? emaskA : emaskB;
      edge_remap<<<nb(kE, TPB), TPB, 0, stream>>>(cs, cd, cm, newpos, ns_, nd_, nm_);
      cs = ns_; cd = nd_; cm = nm_;
      hcur = hnext;
      nper = k;
    }
  }

  // --- classifier MLP ---
  cvt_f32_f16<<<nb(kB * kJK, TPB), TPB, 0, stream>>>(jk, jk16, kB * kJK);
  wmma_gemm_f16<EPI_RELU><<<dim3(kHID / 64, gy(kB)), 128, 0, stream>>>(
      jk16, w1T, b1, nullptr, z1, z16a, kB, kHID, kJK);
  wmma_gemm_f16<EPI_RELU><<<dim3(kHID / 64, gy(kB)), 128, 0, stream>>>(
      z16a, w2T, b2, nullptr, z2, nullptr, kB, kHID, kHID);
  final_linear<<<nb(kB * kNCLS, 64), 64, 0, stream>>>(z2, w3, b3, out);
}
